// DynamicConv_52218212385219
// MI455X (gfx1250) — matrix operand order repaired and checked by reading, not compiled, not run on hardware
//
#include <hip/hip_runtime.h>
#include <hip/hip_bf16.h>
#include <math.h>

// ---------------------------------------------------------------------------
// DynamicConv for MI455X (gfx1250, wave32, WMMA f32_16x16x32_f16)
// B=8 T=2048 C=1024 H=16 K=31 hd=64
// GEMMs double-buffer LDS via GLOBAL_LOAD_ASYNC_TO_LDS (ASYNCcnt) when the
// toolchain exposes the gfx1250 async builtins; otherwise sync staging.
// ---------------------------------------------------------------------------

typedef __attribute__((ext_vector_type(16))) _Float16 v16h;
typedef __attribute__((ext_vector_type(8)))  float    v8f;
typedef int v4i __attribute__((vector_size(16)));

#define BB   8
#define TT   2048
#define CC   1024
#define HH   16
#define KW   31
#define HD   64
#define MTOT (BB * TT)     // 16384 tokens
#define NPAD 512           // Wck N padded 496 -> 512
#define AST  40            // LDS row stride in halves (80B: conflict-free b128)

#if defined(__has_builtin)
#if __has_builtin(__builtin_amdgcn_global_load_async_to_lds_b128) && \
    __has_builtin(__builtin_amdgcn_s_wait_asynccnt)
#define USE_ASYNC 1
#endif
#endif
#ifndef USE_ASYNC
#define USE_ASYNC 0
#endif

union F16Frag { v16h v; uint4 q[2]; };

// 16B global->LDS copy: async DMA into LDS when available, else sync.
__device__ __forceinline__ void cp16(_Float16* l, const _Float16* g) {
#if USE_ASYNC
  __builtin_amdgcn_global_load_async_to_lds_b128(
      (__attribute__((address_space(1))) v4i*)(uintptr_t)g,
      (__attribute__((address_space(3))) v4i*)(uintptr_t)l, 0, 0);
#else
  *(uint4*)l = *(const uint4*)g;
#endif
}

// ------------------------- 1) LayerNorm + cast f16 -------------------------
__global__ __launch_bounds__(256)
void ln_cast_kernel(const float* __restrict__ x,
                    const float* __restrict__ gamma,
                    const float* __restrict__ beta,
                    _Float16* __restrict__ hA) {
  const int row = blockIdx.x;            // token
  const int tid = threadIdx.x;           // 256 threads * 4 floats = 1024
  const float4 v = ((const float4*)(x + (size_t)row * CC))[tid];
  __shared__ float s1[256], s2[256];
  s1[tid] = v.x + v.y + v.z + v.w;
  s2[tid] = v.x * v.x + v.y * v.y + v.z * v.z + v.w * v.w;
  __syncthreads();
  for (int s = 128; s > 0; s >>= 1) {
    if (tid < s) { s1[tid] += s1[tid + s]; s2[tid] += s2[tid + s]; }
    __syncthreads();
  }
  const float mu   = s1[0] * (1.0f / CC);
  const float var  = s2[0] * (1.0f / CC) - mu * mu;
  const float rstd = rsqrtf(var + 1e-5f);
  const int c = tid * 4;
  const float vv[4] = {v.x, v.y, v.z, v.w};
  union { _Float16 h[4]; uint2 u; } pk;
  #pragma unroll
  for (int i = 0; i < 4; i++)
    pk.h[i] = (_Float16)((vv[i] - mu) * rstd * gamma[c + i] + beta[c + i]);
  *(uint2*)(hA + (size_t)row * CC + c) = pk.u;
}

// ------------------- 2) cast + transpose weights to [N][K] f16 -------------
__global__ __launch_bounds__(256)
void cast_transpose_kernel(const float* __restrict__ src,
                           _Float16* __restrict__ dst,
                           int cols, int colsPad) {
  const int idx = blockIdx.x * 256 + threadIdx.x;   // over colsPad*CC
  if (idx >= colsPad * CC) return;
  const int n = idx >> 10;         // / CC
  const int c = idx & (CC - 1);
  dst[(size_t)n * CC + c] = (n < cols) ? (_Float16)src[(size_t)c * cols + n]
                                       : (_Float16)0.0f;
}

// ---------------- 3) GEMM1 (h @ Wk + bk) fused with GLU --------------------
// Block tile: M=128, paired N=64 ("a") + N=64 ("gte"), K stepped by 32,
// double-buffered LDS, async staging. 8 waves; wave w owns M-subtile w.
__global__ __launch_bounds__(256)
void gemm1_glu_kernel(const _Float16* __restrict__ hA,    // [16384][1024]
                      const _Float16* __restrict__ WkT,   // [2048][1024]
                      const float*    __restrict__ bk,    // [2048]
                      _Float16*       __restrict__ kf16)  // [16384][1024]
{
  __shared__ __align__(16) _Float16 Al[2][128 * AST];
  __shared__ __align__(16) _Float16 Bl[2][128 * AST];
  const int tid  = threadIdx.x;
  const int lane = tid & 31;
  const int wv   = tid >> 5;                 // 0..7
  const int m0   = blockIdx.y * 128;
  const int n0   = blockIdx.x * 64;

  v8f acc[8];
  const v8f vzero = {0.f,0.f,0.f,0.f,0.f,0.f,0.f,0.f};
  #pragma unroll
  for (int j = 0; j < 8; j++) acc[j] = vzero;

  auto stage = [&](int kk, int buf) {
    #pragma unroll
    for (int it = 0; it < 2; it++) {
      const int ch = tid + it * 256;         // 512 chunks of 16B per tile
      const int r  = ch >> 2;
      const int o2 = (ch & 3) * 8;
      cp16(&Al[buf][r * AST + o2], hA + (size_t)(m0 + r) * CC + kk + o2);
      const int gcol = (r < 64) ? (n0 + r) : (CC + n0 + r - 64);
      cp16(&Bl[buf][r * AST + o2], WkT + (size_t)gcol * CC + kk + o2);
    }
  };

  stage(0, 0);
  const int NSLAB = CC / 32;                 // 32
  for (int i = 0; i < NSLAB; ++i) {
    const int cur = i & 1;
    if (i + 1 < NSLAB) {
      stage((i + 1) * 32, cur ^ 1);          // prefetch next slab (other buf)
#if USE_ASYNC
      __builtin_amdgcn_s_wait_asynccnt(4);   // retire current slab only
#endif
    } else {
#if USE_ASYNC
      __builtin_amdgcn_s_wait_asynccnt(0);
#endif
    }
    __syncthreads();

    // A fragment: lane<16 -> M=lane,    K = {0..7,16..23}
    //             lane>=16 -> M=lane-16, K = {8..15,24..31}
    const _Float16* Ab = Al[cur];
    const _Float16* Bb = Bl[cur];
    const int mloc = wv * 16 + (lane & 15);
    const int ksel = (lane < 16) ? 0 : 8;
    F16Frag Af;
    Af.q[0] = *(const uint4*)(Ab + mloc * AST + ksel);
    Af.q[1] = *(const uint4*)(Ab + mloc * AST + ksel + 16);

    const int kselB = (lane < 16) ? 0 : 8;   // B: lane<16 -> K={0..7,16..23}
    #pragma unroll
    for (int j = 0; j < 8; j++) {
      const int nloc = j * 16 + (lane & 15);
      F16Frag Bf;
      Bf.q[0] = *(const uint4*)(Bb + nloc * AST + kselB);
      Bf.q[1] = *(const uint4*)(Bb + nloc * AST + kselB + 16);
      acc[j] = __builtin_amdgcn_wmma_f32_16x16x32_f16(
          false, Af.v, false, Bf.v, (short)0, acc[j], false, false);
    }
    __syncthreads();                         // all reads of buf[cur] done
  }

  // GLU epilogue: k = (a + ba) * sigmoid(g + bg); C layout: lane%16 = N,
  // VGPR e -> M = e + 8*(lane>=16).
  const int nlane = lane & 15;
  const int mhi   = (lane >> 4) * 8;
  #pragma unroll
  for (int j = 0; j < 4; j++) {
    const int col = n0 + j * 16 + nlane;
    const float ba = bk[col];
    const float bg = bk[CC + col];
    #pragma unroll
    for (int e = 0; e < 8; e++) {
      const float a = acc[j][e] + ba;
      const float g = acc[j + 4][e] + bg;
      const float kv = a / (1.0f + __expf(-g));
      const int row = m0 + wv * 16 + mhi + e;
      kf16[(size_t)row * CC + col] = (_Float16)kv;
    }
  }
}

// ------------------- 4) GEMM2 (k @ Wck + bck) -> logits f32 ----------------
__global__ __launch_bounds__(256)
void gemm2_kernel(const _Float16* __restrict__ kf16,  // [16384][1024]
                  const _Float16* __restrict__ WckT,  // [512][1024] (padded)
                  const float*    __restrict__ bck,   // [496]
                  float*          __restrict__ wlog)  // [16384][496]
{
  __shared__ __align__(16) _Float16 Al[2][128 * AST];
  __shared__ __align__(16) _Float16 Bl[2][64 * AST];
  const int tid  = threadIdx.x;
  const int lane = tid & 31;
  const int wv   = tid >> 5;
  const int m0   = blockIdx.y * 128;
  const int n0   = blockIdx.x * 64;     // 0..448

  v8f acc[4];
  const v8f vzero = {0.f,0.f,0.f,0.f,0.f,0.f,0.f,0.f};
  #pragma unroll
  for (int j = 0; j < 4; j++) acc[j] = vzero;

  auto stage = [&](int kk, int buf) {
    #pragma unroll
    for (int it = 0; it < 2; it++) {
      const int ch = tid + it * 256;
      const int r  = ch >> 2;
      const int o2 = (ch & 3) * 8;
      cp16(&Al[buf][r * AST + o2], kf16 + (size_t)(m0 + r) * CC + kk + o2);
    }
    {
      const int r  = tid >> 2;          // 0..63
      const int o2 = (tid & 3) * 8;
      cp16(&Bl[buf][r * AST + o2], WckT + (size_t)(n0 + r) * CC + kk + o2);
    }
  };

  stage(0, 0);
  const int NSLAB = CC / 32;
  for (int i = 0; i < NSLAB; ++i) {
    const int cur = i & 1;
    if (i + 1 < NSLAB) {
      stage((i + 1) * 32, cur ^ 1);
#if USE_ASYNC
      __builtin_amdgcn_s_wait_asynccnt(3);
#endif
    } else {
#if USE_ASYNC
      __builtin_amdgcn_s_wait_asynccnt(0);
#endif
    }
    __syncthreads();

    const _Float16* Ab = Al[cur];
    const _Float16* Bb = Bl[cur];
    const int mloc = wv * 16 + (lane & 15);
    const int ksel = (lane < 16) ? 0 : 8;
    F16Frag Af;
    Af.q[0] = *(const uint4*)(Ab + mloc * AST + ksel);
    Af.q[1] = *(const uint4*)(Ab + mloc * AST + ksel + 16);

    const int kselB = (lane < 16) ? 0 : 8;
    #pragma unroll
    for (int j = 0; j < 4; j++) {
      const int nloc = j * 16 + (lane & 15);
      F16Frag Bf;
      Bf.q[0] = *(const uint4*)(Bb + nloc * AST + kselB);
      Bf.q[1] = *(const uint4*)(Bb + nloc * AST + kselB + 16);
      acc[j] = __builtin_amdgcn_wmma_f32_16x16x32_f16(
          false, Af.v, false, Bf.v, (short)0, acc[j], false, false);
    }
    __syncthreads();
  }

  const int nlane = lane & 15;
  const int mhi   = (lane >> 4) * 8;
  #pragma unroll
  for (int j = 0; j < 4; j++) {
    const int col = n0 + j * 16 + nlane;
    if (col < HH * KW) {
      const float bb = bck[col];
      #pragma unroll
      for (int e = 0; e < 8; e++) {
        const int row = m0 + wv * 16 + mhi + e;
        wlog[(size_t)row * (HH * KW) + col] = acc[j][e] + bb;
      }
    }
  }
}

// ------------------- 5) softmax over K=31 per (token, head) ----------------
__global__ __launch_bounds__(256)
void softmax_kernel(float* __restrict__ wlog) {
  const int idx = blockIdx.x * 256 + threadIdx.x;   // token*16 + head
  if (idx >= MTOT * HH) return;
  float* p = wlog + (size_t)(idx >> 4) * (HH * KW) + (idx & 15) * KW;
  float m = -INFINITY;
  #pragma unroll
  for (int i = 0; i < KW; i++) m = fmaxf(m, p[i]);
  float s = 0.0f;
  #pragma unroll
  for (int i = 0; i < KW; i++) s += __expf(p[i] - m);
  const float inv = 1.0f / s;
  #pragma unroll
  for (int i = 0; i < KW; i++) p[i] = __expf(p[i] - m) * inv;
}

// -------- 6) depthwise dynamic conv + conv_bias + residual (LDS stencil) ---
__global__ __launch_bounds__(256)
void dynconv_kernel(const _Float16* __restrict__ kf16,  // [B*T][C] f16
                    const float*    __restrict__ wsm,   // [B*T][496] softmaxed
                    const float*    __restrict__ x,
                    const float*    __restrict__ conv_bias,
                    float*          __restrict__ out) {
  const int b  = blockIdx.x >> 4;
  const int h  = blockIdx.x & 15;
  const int t0 = blockIdx.y * 128;
  __shared__ __align__(16) _Float16 kt[(128 + KW - 1) * HD];  // 158x64 halves
  __shared__ float wt[128 * KW];
  const int tid = threadIdx.x;

  // stage k stencil rows t0-15 .. t0+142 (zero-padded at sequence edges)
  for (int c = tid; c < (128 + KW - 1) * (HD / 8); c += 256) {
    const int r   = c >> 3;
    const int off = (c & 7) * 8;
    const int tp  = t0 - (KW / 2) + r;
    uint4 val = {0u, 0u, 0u, 0u};
    if (tp >= 0 && tp < TT)
      val = *(const uint4*)(kf16 + (size_t)(b * TT + tp) * CC + h * HD + off);
    *(uint4*)(kt + r * HD + off) = val;
  }
  // stage per-token kernels
  for (int i = tid; i < 128 * KW; i += 256) {
    const int tl = i / KW;
    const int kk = i - tl * KW;
    wt[i] = wsm[(size_t)(b * TT + t0 + tl) * (HH * KW) + h * KW + kk];
  }
  __syncthreads();

  const int d    = tid & 63;
  const int trow = tid >> 6;
  const int col  = h * HD + d;
  const float cb = conv_bias[col];
  for (int tl = trow; tl < 128; tl += 4) {
    float acc = 0.0f;
    #pragma unroll
    for (int i = 0; i < KW; i++)
      acc += wt[tl * KW + i] * (float)kt[(tl + i) * HD + d];
    const size_t g = (size_t)(b * TT + t0 + tl) * CC + col;
    out[g] = acc + cb + x[g];
  }
}

// ---------------------------------------------------------------------------
extern "C" void kernel_launch(void* const* d_in, const int* in_sizes, int n_in,
                              void* d_out, int out_size, void* d_ws, size_t ws_size,
                              hipStream_t stream) {
  (void)in_sizes; (void)n_in; (void)out_size; (void)ws_size;
  const float* x         = (const float*)d_in[0];
  const float* Wk        = (const float*)d_in[1];
  const float* bk        = (const float*)d_in[2];
  const float* Wck       = (const float*)d_in[3];
  const float* bck       = (const float*)d_in[4];
  const float* gamma     = (const float*)d_in[5];
  const float* beta      = (const float*)d_in[6];
  const float* conv_bias = (const float*)d_in[7];
  float* out = (float*)d_out;

  char* ws = (char*)d_ws;
  size_t off = 0;
  auto take = [&](size_t bytes) -> char* {
    char* p = ws + off;
    off = (off + bytes + 255) & ~(size_t)255;
    return p;
  };
  _Float16* hA   = (_Float16*)take((size_t)MTOT * CC * 2);   //  32 MiB
  _Float16* WkT  = (_Float16*)take((size_t)2048 * CC * 2);   //   4 MiB
  _Float16* WckT = (_Float16*)take((size_t)NPAD * CC * 2);   //   1 MiB
  _Float16* kf16 = (_Float16*)take((size_t)MTOT * CC * 2);   //  32 MiB
  float*    wlog = (float*)take((size_t)MTOT * HH * KW * 4); //  31 MiB

  ln_cast_kernel<<<MTOT, 256, 0, stream>>>(x, gamma, beta, hA);
  cast_transpose_kernel<<<(2048 * CC) / 256, 256, 0, stream>>>(Wk, WkT, 2048, 2048);
  cast_transpose_kernel<<<(NPAD * CC) / 256, 256, 0, stream>>>(Wck, WckT, HH * KW, NPAD);
  gemm1_glu_kernel<<<dim3(16, MTOT / 128), 256, 0, stream>>>(hA, WkT, bk, kf16);
  gemm2_kernel<<<dim3(NPAD / 64, MTOT / 128), 256, 0, stream>>>(kf16, WckT, bck, wlog);
  softmax_kernel<<<(MTOT * HH) / 256, 256, 0, stream>>>(wlog);
  dynconv_kernel<<<dim3(BB * HH, TT / 128), 256, 0, stream>>>(kf16, wlog, x, conv_bias, out);
}
